// SimpleLSTM_65833258713577
// MI455X (gfx1250) — compile-verified, matrix operand order checked by reading, not run on hardware
//
#include <hip/hip_runtime.h>
#include <hip/hip_bf16.h>
#include <math.h>

// ---------------------------------------------------------------------------
// SimpleLSTM for MI455X (gfx1250):
//  - bf16 WMMA input-projection GEMM, double-buffered LDS, L2-blocked over T
//  - 16-WG persistent recurrence with LDS-resident Wh (padded, conflict-free),
//    per-step async global->LDS h broadcast, grid ticket barrier per step.
// ---------------------------------------------------------------------------

typedef __attribute__((ext_vector_type(16))) __bf16 v16bf;
typedef __attribute__((ext_vector_type(8)))  float  v8f;

#define B_   64
#define T_   2048
#define I_   512
#define H_   512
#define G_   2048          // 4*H
#define TCHUNK 128         // gates chunk = 128*64*2048*4 B = 64 MiB (fits 192MB L2)
#define NWG_SCAN 16

// Padded LDS strides (conflict-free fragment loads)
#define WH_STRIDE 520      // bf16 elems per Wh row   (1040 B, 260 dw % 64 == 4)
#define HS_STRIDE 520      // bf16 elems per h row
#define GS_STRIDE 132      // f32 elems per gate-staging row
#define AB_STRIDE 40       // bf16 elems per GEMM tile row (80 B, 20 dw)

#define WHS_BYTES (128 * WH_STRIDE * 2)     // 133120
#define HS_OFF    WHS_BYTES
#define HS_BYTES  (B_ * HS_STRIDE * 2)      // 66560
#define GS_OFF    (HS_OFF + HS_BYTES)       // 199680
#define GS_BYTES  (B_ * GS_STRIDE * 4)      // 33792
#define SCAN_LDS  (GS_OFF + GS_BYTES)       // 233472 <= 320K WGP LDS

__device__ __forceinline__ float fast_sigmoid(float x) {
  return 1.0f / (1.0f + __expf(-x));
}
__device__ __forceinline__ float fast_tanh(float x) {
  return 1.0f - 2.0f / (__expf(2.0f * x) + 1.0f);
}

// 32B fragment from LDS as two 16B reads (rows are 16B-aligned, not 32B).
__device__ __forceinline__ v16bf ld_frag_lds(const __bf16* p) {
  union { uint4 q[2]; v16bf v; } u;
  u.q[0] = *(const uint4*)(p);
  u.q[1] = *(const uint4*)(p + 8);
  return u.v;
}

// ---------------------------------------------------------------------------
// init: zero barrier counter, h ping-pong (h0=0), c state (c0=0)
// ---------------------------------------------------------------------------
__global__ void lstm_init(unsigned* __restrict__ bar,
                          __bf16* __restrict__ hbuf,   // [2][B_*H_]
                          float*  __restrict__ cbuf) { // [B_*H_]
  int i = blockIdx.x * blockDim.x + threadIdx.x;
  if (i == 0) *bar = 0u;
  if (i < 2 * B_ * H_) hbuf[i] = (__bf16)0.0f;
  if (i < B_ * H_)     cbuf[i] = 0.0f;
}

// ---------------------------------------------------------------------------
// Phase 1: gates[(t*64+b), g] = x[b, t0+t, :] . Wx[g, :] + (bx[g]+bh[g])
// 128(M) x 128(N) per WG; K staged 32 at a time, double-buffered bf16 LDS.
// Wave = (row-tile pair) x (4 col-tiles): 2 A frags + 4 B frags -> 8 WMMAs.
// ---------------------------------------------------------------------------
__global__ void __launch_bounds__(256)
lstm_gates_gemm(const float* __restrict__ x,
                const float* __restrict__ Wx,
                const float* __restrict__ bx,
                const float* __restrict__ bh,
                float* __restrict__ gates,   // [TCHUNK*64, 2048]
                int t0) {
  __shared__ __attribute__((aligned(16))) __bf16 As[2][128 * AB_STRIDE];
  __shared__ __attribute__((aligned(16))) __bf16 Bs[2][128 * AB_STRIDE];

  const int mblock = blockIdx.x;
  const int nblock = blockIdx.y;
  const int tid  = threadIdx.x;
  const int wv   = tid >> 5;
  const int lane = tid & 31;

  const int frow = lane & 15;            // M (A) / N (B) within tile
  const int fkb  = (lane >> 4) * 16;     // K half (contiguous k = 16h+s, same for A and B)
  const int rg   = wv & 3;               // row-tile pair: tiles 2rg, 2rg+1
  const int cgp  = wv >> 2;              // col tiles cgp*4 .. cgp*4+3

  // staging registers (prefetch for next K slice)
  float4 ra[4], rb[4];
  const int sr  = tid >> 3;              // staging row base (8 thr/row)
  const int sc4 = tid & 7;               // float4 column

  auto issue_loads = [&](int k0) {
    #pragma unroll
    for (int i = 0; i < 4; ++i) {
      int r = sr + i * 32;               // tile row 0..127
      int R = mblock * 128 + r;
      int tl = R >> 6, b = R & 63;
      ra[i] = *(const float4*)(x + ((size_t)b * T_ + (t0 + tl)) * I_ + k0 + sc4 * 4);
      int gc = nblock * 128 + r;
      rb[i] = *(const float4*)(Wx + (size_t)gc * I_ + k0 + sc4 * 4);
    }
    // Pin the scheduler: all 8 global loads must be *issued* here, before the
    // WMMA section; the single s_wait_loadcnt lands at the cvt block below.
    __builtin_amdgcn_sched_barrier(0);
  };
  auto store_stage = [&](int buf) {
    #pragma unroll
    for (int i = 0; i < 4; ++i) {
      int r = sr + i * 32;
      __bf16* da = &As[buf][r * AB_STRIDE + sc4 * 4];
      da[0] = (__bf16)ra[i].x; da[1] = (__bf16)ra[i].y;
      da[2] = (__bf16)ra[i].z; da[3] = (__bf16)ra[i].w;
      __bf16* db = &Bs[buf][r * AB_STRIDE + sc4 * 4];
      db[0] = (__bf16)rb[i].x; db[1] = (__bf16)rb[i].y;
      db[2] = (__bf16)rb[i].z; db[3] = (__bf16)rb[i].w;
    }
  };

  issue_loads(0);
  store_stage(0);
  __syncthreads();

  v8f acc[2][4] = {};

  for (int ki = 0; ki < 16; ++ki) {
    const int cur = ki & 1;
    if (ki < 15) issue_loads((ki + 1) * 32);   // overlap global with WMMA

    v16bf a0 = ld_frag_lds(&As[cur][((2 * rg    ) * 16 + frow) * AB_STRIDE + fkb]);
    v16bf a1 = ld_frag_lds(&As[cur][((2 * rg + 1) * 16 + frow) * AB_STRIDE + fkb]);
    #pragma unroll
    for (int q = 0; q < 4; ++q) {
      int ct = cgp * 4 + q;
      v16bf bf = ld_frag_lds(&Bs[cur][(ct * 16 + frow) * AB_STRIDE + fkb]);
      acc[0][q] = __builtin_amdgcn_wmma_f32_16x16x32_bf16(
          false, a0, false, bf, (short)0, acc[0][q], false, false);
      acc[1][q] = __builtin_amdgcn_wmma_f32_16x16x32_bf16(
          false, a1, false, bf, (short)0, acc[1][q], false, false);
    }
    if (ki < 15) store_stage(cur ^ 1);
    __syncthreads();
  }

  // Epilogue: + (bx+bh), store f32 gates chunk.
  const int n15 = lane & 15;
  const int mo  = (lane >> 4) * 8;       // C/D layout: lanes 16-31 hold M+8
  #pragma unroll
  for (int q = 0; q < 4; ++q) {
    int gc = nblock * 128 + (cgp * 4 + q) * 16 + n15;
    float bias = bx[gc] + bh[gc];
    #pragma unroll
    for (int rr = 0; rr < 2; ++rr) {
      #pragma unroll
      for (int r = 0; r < 8; ++r) {
        int R = mblock * 128 + (2 * rg + rr) * 16 + r + mo;
        gates[(size_t)R * G_ + gc] = acc[rr][q][r] + bias;
      }
    }
  }
}

// ---------------------------------------------------------------------------
// Grid-wide ticket barrier (16 co-resident WGs; one per WGP given 228KB LDS).
// Counter is monotonic; valid for any start value that's a multiple of NWG.
// All threads fence on exit so both CUs of each WGP perform the acquire.
// ---------------------------------------------------------------------------
__device__ __forceinline__ void grid_barrier(unsigned* bar) {
  __threadfence();
  __syncthreads();
  if (threadIdx.x == 0) {
    unsigned ticket = __hip_atomic_fetch_add(bar, 1u, __ATOMIC_ACQ_REL,
                                             __HIP_MEMORY_SCOPE_AGENT);
    unsigned target = (ticket / NWG_SCAN + 1u) * NWG_SCAN;
    while (__hip_atomic_load(bar, __ATOMIC_ACQUIRE,
                             __HIP_MEMORY_SCOPE_AGENT) < target) {
      __builtin_amdgcn_s_sleep(1);
    }
  }
  __syncthreads();
  __threadfence();
}

// ---------------------------------------------------------------------------
// Phase 2: persistent recurrence over TCHUNK steps.
// WG w owns h-columns [w*32, w*32+32) i.e. gate columns g*512 + w*32 + j.
// Wh slice (128 x 520 bf16) in LDS for all steps; h broadcast per step via
// async global->LDS copy; c state in registers for the whole chunk.
// ---------------------------------------------------------------------------
__global__ void __launch_bounds__(256)
lstm_scan(const float* __restrict__ gates,   // [TCHUNK*64, 2048]
          const float* __restrict__ Wh,      // [2048, 512] f32
          float* __restrict__ out,           // [64, 2048, 512] (+ tails)
          __bf16* __restrict__ hbuf,         // [2][64*512]
          float*  __restrict__ cbuf,         // [64*512]
          unsigned* __restrict__ bar,
          int t0) {
  extern __shared__ __attribute__((aligned(128))) char smem[];
  __bf16* WhS = (__bf16*)smem;
  __bf16* hS  = (__bf16*)(smem + HS_OFF);
  float*  gS  = (float*)(smem + GS_OFF);

  const int w    = blockIdx.x;        // 0..15
  const int tid  = threadIdx.x;
  const int wv   = tid >> 5;
  const int lane = tid & 31;

  // Stage this WG's Wh slice (f32 -> bf16), padded stride.
  for (int i = tid; i < 128 * (H_ / 4); i += 256) {
    int lc = i >> 7;
    int c4 = i & 127;
    int gr = ((lc >> 5) * H_) + w * 32 + (lc & 31);
    const float4 v = *(const float4*)(Wh + (size_t)gr * H_ + c4 * 4);
    __bf16* d = WhS + lc * WH_STRIDE + c4 * 4;
    d[0] = (__bf16)v.x; d[1] = (__bf16)v.y; d[2] = (__bf16)v.z; d[3] = (__bf16)v.w;
  }
  __syncthreads();

  const int frow = lane & 15;
  const int fkb  = (lane >> 4) * 16;
  const int n15  = lane & 15;
  const int mo   = (lane >> 4) * 8;
  const int rt   = wv & 3;            // row tile (16 of 64 batch rows)
  const int cg   = wv >> 2;           // col-tile group: tiles cg*4 .. cg*4+3

  // c state lives in registers for the whole chunk (static thread<->elem map)
  float creg[8];
  #pragma unroll
  for (int e = 0; e < 8; ++e) {
    int idx = tid + e * 256;
    creg[e] = cbuf[(idx >> 5) * H_ + w * 32 + (idx & 31)];
  }

  for (int tt = 0; tt < TCHUNK; ++tt) {
    const int t = t0 + tt;
    const __bf16* hin = hbuf + (size_t)(t & 1) * (B_ * H_);

    // (1) Async broadcast copy: h (64x512 bf16, dense) -> hS (padded rows).
    #pragma unroll
    for (int q = 0; q < 16; ++q) {
      int i = tid + q * 256;               // 0..4095 16B packets
      int r = i >> 6, part = i & 63;
      unsigned ldsa = (unsigned)(HS_OFF + r * (HS_STRIDE * 2) + part * 16);
      unsigned long long ga =
          (unsigned long long)(uintptr_t)((const char*)hin + r * 1024 + part * 16);
      asm volatile("global_load_async_to_lds_b128 %0, %1, off"
                   :: "v"(ldsa), "v"(ga) : "memory");
    }

    // (2) Prefetch this step's gates_x slice while the copy flies.
    float gx[4][8];
    #pragma unroll
    for (int e = 0; e < 8; ++e) {
      int idx = tid + e * 256;
      int m = idx >> 5, j = idx & 31;
      const float* g = gates + ((size_t)tt * B_ + m) * G_ + w * 32 + j;
      gx[0][e] = g[0];
      gx[1][e] = g[512];
      gx[2][e] = g[1024];
      gx[3][e] = g[1536];
    }
    // Pin issue of the gates loads ahead of the async wait below.
    __builtin_amdgcn_sched_barrier(0);

    asm volatile("s_wait_asynccnt 0x0" ::: "memory");
    __syncthreads();

    // (3) gates_h tile: (64 x 128) = h . WhS^T, K in steps of 32, all LDS.
    v8f acc[4] = {};
    for (int k0 = 0; k0 < H_; k0 += 32) {
      v16bf a = ld_frag_lds(hS + (rt * 16 + frow) * HS_STRIDE + k0 + fkb);
      #pragma unroll
      for (int q = 0; q < 4; ++q) {
        int ctl = cg * 4 + q;
        v16bf bf = ld_frag_lds(WhS + (ctl * 16 + frow) * WH_STRIDE + k0 + fkb);
        acc[q] = __builtin_amdgcn_wmma_f32_16x16x32_bf16(
            false, a, false, bf, (short)0, acc[q], false, false);
      }
    }
    // (4) Spill accumulators to LDS so threads can gather i/f/g/o.
    #pragma unroll
    for (int q = 0; q < 4; ++q) {
      int ctl = cg * 4 + q;
      #pragma unroll
      for (int r = 0; r < 8; ++r) {
        gS[(rt * 16 + r + mo) * GS_STRIDE + ctl * 16 + n15] = acc[q][r];
      }
    }
    __syncthreads();

    // (5) Elementwise LSTM cell for the WG's 64x32 slice.
    #pragma unroll
    for (int e = 0; e < 8; ++e) {
      int idx = tid + e * 256;
      int m = idx >> 5;
      int j = idx & 31;
      float iv = gS[m * GS_STRIDE +  0 + j] + gx[0][e];
      float fv = gS[m * GS_STRIDE + 32 + j] + gx[1][e];
      float gv = gS[m * GS_STRIDE + 64 + j] + gx[2][e];
      float ov = gS[m * GS_STRIDE + 96 + j] + gx[3][e];
      iv = fast_sigmoid(iv);
      fv = fast_sigmoid(fv);
      gv = fast_tanh(gv);
      ov = fast_sigmoid(ov);
      float c = fv * creg[e] + iv * gv;
      creg[e] = c;
      float h = ov * fast_tanh(c);
      int ci = m * H_ + w * 32 + j;
      out[((size_t)m * T_ + t) * H_ + w * 32 + j] = h;
      hbuf[(size_t)((t + 1) & 1) * (B_ * H_) + ci] = (__bf16)h;
      if (t == T_ - 1) {
        out[(size_t)B_ * T_ * H_ + ci]           = h;   // h_n
        out[(size_t)B_ * T_ * H_ + B_ * H_ + ci] = c;   // c_n
      }
    }
    // (6) Publish h_t to all WGs before anyone starts step t+1.
    grid_barrier(bar);
  }

  // Persist c for the next chunk launch.
  #pragma unroll
  for (int e = 0; e < 8; ++e) {
    int idx = tid + e * 256;
    cbuf[(idx >> 5) * H_ + w * 32 + (idx & 31)] = creg[e];
  }
}

// ---------------------------------------------------------------------------
// Launcher
// ---------------------------------------------------------------------------
extern "C" void kernel_launch(void* const* d_in, const int* in_sizes, int n_in,
                              void* d_out, int out_size, void* d_ws, size_t ws_size,
                              hipStream_t stream) {
  const float* x  = (const float*)d_in[0];
  const float* Wx = (const float*)d_in[1];
  const float* bx = (const float*)d_in[2];
  const float* Wh = (const float*)d_in[3];
  const float* bh = (const float*)d_in[4];
  float* out = (float*)d_out;

  char* ws = (char*)d_ws;
  unsigned* bar  = (unsigned*)ws;                                  // 1 KB reserved
  __bf16*   hbuf = (__bf16*)(ws + 1024);                           // 128 KB
  float*    cbuf = (float*)(ws + 1024 + 2 * B_ * H_ * 2);          // 128 KB
  float*    gates = (float*)(ws + 1024 + 2 * B_ * H_ * 2 + B_ * H_ * 4); // 64 MiB

  lstm_init<<<256, 256, 0, stream>>>(bar, hbuf, cbuf);

  for (int chunk = 0; chunk < T_ / TCHUNK; ++chunk) {
    int t0 = chunk * TCHUNK;
    lstm_gates_gemm<<<dim3(TCHUNK * B_ / 128, G_ / 128), 256, 0, stream>>>(
        x, Wx, bx, bh, gates, t0);
    lstm_scan<<<NWG_SCAN, 256, SCAN_LDS, stream>>>(
        gates, Wh, out, hbuf, cbuf, bar, t0);
  }
}
